// CrossAttention_24464133718265
// MI455X (gfx1250) — matrix operand order repaired and checked by reading, not compiled, not run on hardware
//
#include <hip/hip_runtime.h>
#include <hip/hip_bf16.h>
#include <stdint.h>

typedef __attribute__((ext_vector_type(16))) __bf16 v16bf;
typedef __attribute__((ext_vector_type(8)))  __bf16 v8bf;
typedef __attribute__((ext_vector_type(4)))  __bf16 v4bf;
typedef __attribute__((ext_vector_type(8)))  float  v8f;

union BF16x16 { v16bf v; v8bf h[2]; };

#define IN_DIM   1024
#define D_DIM    128     // KD*H = VD*H
#define M_TILE   32      // rows of B per block
#define KC       64      // K-chunk for QKV GEMMs
#define NC       64      // N-chunk for output GEMM

// ws layout (bf16): wsW [16 kchunks][3 mats][128 n][64 k]  = 393216 elems
//                   wsWo [1024 n][128 k]                   = 131072 elems
#define WSW_ELEMS  (16 * 3 * 128 * 64)
#define WSWO_ELEMS (1024 * 128)
#define WS_BYTES   ((size_t)(WSW_ELEMS + WSWO_ELEMS) * 2)

// ---- CDNA5 async global->LDS copy (ASYNCcnt-tracked) ----
__device__ __forceinline__ void async_b128(uint32_t lds_addr, uint32_t voff,
                                           const void* sbase) {
  asm volatile("global_load_async_to_lds_b128 %0, %1, %2"
               :: "v"(lds_addr), "v"(voff), "s"(sbase) : "memory");
}
__device__ __forceinline__ void wait_async0() {
#if __has_builtin(__builtin_amdgcn_s_wait_asynccnt)
  __builtin_amdgcn_s_wait_asynccnt(0);
#else
  asm volatile("s_wait_asynccnt 0x0" ::: "memory");
#endif
}

// A-frag (16x32 bf16, MxK) from row-major LDS tile [rows][stride]
__device__ __forceinline__ v16bf load_a_frag(const __bf16* base, int stride,
                                             int row0, int kb, int lane) {
  int lo = lane & 15, hi = lane >> 4;
  const __bf16* p = base + (row0 + lo) * stride + kb + hi * 8;
  BF16x16 f;
  f.h[0] = *(const v8bf*)(p);
  f.h[1] = *(const v8bf*)(p + 16);
  return f.v;
}

// B-frag (32x16 bf16, KxN) from K-contiguous (transposed) LDS tile [N][stride]
__device__ __forceinline__ v16bf load_b_frag(const __bf16* baseT, int stride,
                                             int col0, int kb, int lane) {
  int lo = lane & 15, hi = lane >> 4;
  const __bf16* p = baseT + (col0 + lo) * stride + kb + hi * 8;
  BF16x16 f;
  f.h[0] = *(const v8bf*)(p);
  f.h[1] = *(const v8bf*)(p + 16);
  return f.v;
}

// One-shot weight convert/transpose into chunk-tiled bf16 images.
__global__ __launch_bounds__(256)
void prep_weights(const float* __restrict__ Wq, const float* __restrict__ Wk,
                  const float* __restrict__ Wv, const float* __restrict__ Wo,
                  unsigned short* __restrict__ wsW_, unsigned short* __restrict__ wsWo_) {
  __bf16* wsW  = (__bf16*)wsW_;
  __bf16* wsWo = (__bf16*)wsWo_;
  int idx = blockIdx.x * 256 + threadIdx.x;          // 524288 total
  if (idx < WSW_ELEMS) {
    int cm  = idx >> 13;          // /(128*64)
    int rem = idx & 8191;
    int n   = rem >> 6;
    int k   = rem & 63;
    int c   = cm / 3;
    int mat = cm % 3;
    const float* W = (mat == 0) ? Wq : (mat == 1) ? Wk : Wv;
    wsW[idx] = (__bf16)W[(size_t)(c * 64 + k) * D_DIM + n];
  } else {
    int e = idx - WSW_ELEMS;      // 131072: WoT[n][k]
    int n = e >> 7;
    int k = e & 127;
    wsWo[e] = (__bf16)Wo[(size_t)k * IN_DIM + n];
  }
}

__global__ __launch_bounds__(256)
void xattn_fused(const float* __restrict__ x1, const float* __restrict__ x2,
                 const float* __restrict__ Wq, const float* __restrict__ Wk,
                 const float* __restrict__ Wv, const float* __restrict__ Wo,
                 const float* __restrict__ bo, float* __restrict__ out,
                 const unsigned short* __restrict__ wsW_,
                 const unsigned short* __restrict__ wsWo_, int useWs) {
  // LDS plan (56 KB, phase-overlapped):
  //  [0,8K)  sX  : bf16 x1/x2 chunks [2][32][64]
  //  [8K,56K) sW : bf16 W^T chunks [3][128][64] (A)
  //           sQKV [3][32][128] 24K | sO [32][128] 8K | sWo [64][128] 16K (B/C)
  __shared__ alignas(16) unsigned char smem[57344];
  __bf16* sX   = (__bf16*)smem;
  __bf16* sW   = (__bf16*)(smem + 8192);
  __bf16* sQKV = (__bf16*)(smem + 8192);
  __bf16* sO   = sQKV + 3 * 32 * 128;
  __bf16* sWo  = sO + 32 * 128;

  const int tid  = threadIdx.x;
  const int w    = tid >> 5;       // 8 waves (wave32)
  const int lane = tid & 31;
  const int lo   = lane & 15;
  const int hi   = lane >> 4;
  const int row0 = blockIdx.x * M_TILE;
  const int w16  = w * 16;         // this wave's N-column block in phase A

  const __bf16* wsW = (const __bf16*)wsW_;

  // ======== Phase A: Q = x1*Wq, K = x2*Wk, V = x2*Wv ========
  // tile id = w + 8*j: mat = j>>1 (const), tm = j&1 (const), tn = w (uniform)
  v8f acc[6] = {};
  for (int kc = 0; kc < IN_DIM; kc += KC) {
    // stage x1/x2 [32][64] fp32 -> bf16 (coalesced float4 reads)
    #pragma unroll
    for (int i = 0; i < 4; ++i) {
      int idx = tid + i * 256;                 // 1024 float4
      int mat = idx >> 9;
      int rem = idx & 511;
      int row = rem >> 4;
      int c4  = rem & 15;
      const float* src = mat ? x2 : x1;
      float4 xv = *(const float4*)(src + (size_t)(row0 + row) * IN_DIM + kc + c4 * 4);
      v4bf pb;
      pb[0] = (__bf16)xv.x; pb[1] = (__bf16)xv.y;
      pb[2] = (__bf16)xv.z; pb[3] = (__bf16)xv.w;
      *(v4bf*)(sX + mat * (32 * 64) + row * 64 + c4 * 4) = pb;
    }
    // stage W chunks: async copy of pre-tiled bf16 image, else convert fp32
    if (useWs) {
      const __bf16* chunk = wsW + (size_t)(kc >> 6) * (3 * 128 * 64);
      uint32_t ldsBase = (uint32_t)(uintptr_t)sW;
      #pragma unroll
      for (int i = 0; i < 12; ++i) {
        uint32_t boff = (uint32_t)(tid + i * 256) * 16u;   // 3072 x b128
        async_b128(ldsBase + boff, boff, chunk);
      }
    } else {
      for (int i = 0; i < 24; ++i) {
        int idx = tid + i * 256;                 // 6144 float4
        int mat = idx >> 11;
        int rem = idx & 2047;
        int kk  = rem >> 5;
        int n4  = rem & 31;
        const float* ws = (mat == 0) ? Wq : (mat == 1) ? Wk : Wv;
        float4 wv = *(const float4*)(ws + (size_t)(kc + kk) * D_DIM + n4 * 4);
        __bf16* dst = sW + mat * (128 * 64);
        dst[(n4 * 4 + 0) * 64 + kk] = (__bf16)wv.x;
        dst[(n4 * 4 + 1) * 64 + kk] = (__bf16)wv.y;
        dst[(n4 * 4 + 2) * 64 + kk] = (__bf16)wv.z;
        dst[(n4 * 4 + 3) * 64 + kk] = (__bf16)wv.w;
      }
    }
    // prefetch next x chunk while WMMAs run
    if (kc + KC < IN_DIM && tid < 128) {
      int m = tid >> 6, t = tid & 63;
      int prow = t >> 1, half = t & 1;
      const float* src = m ? x2 : x1;
      __builtin_prefetch(src + (size_t)(row0 + prow) * IN_DIM + kc + KC + half * 32, 0, 3);
    }
    if (useWs) wait_async0();
    __syncthreads();
    #pragma unroll
    for (int ks = 0; ks < 2; ++ks) {
      int kb = ks * 32;
      // 4 shared A-frags (x1/x2 x rows 0-15/16-31), 3 B-frags (Wq/Wk/Wv col w16)
      v16bf a10 = load_a_frag(sX,             KC, 0,  kb, lane);
      v16bf a11 = load_a_frag(sX,             KC, 16, kb, lane);
      v16bf a20 = load_a_frag(sX + 32 * 64,   KC, 0,  kb, lane);
      v16bf a21 = load_a_frag(sX + 32 * 64,   KC, 16, kb, lane);
      v16bf bq  = load_b_frag(sW,                KC, w16, kb, lane);
      v16bf bk  = load_b_frag(sW + 128 * 64,     KC, w16, kb, lane);
      v16bf bv  = load_b_frag(sW + 2 * 128 * 64, KC, w16, kb, lane);
      acc[0] = __builtin_amdgcn_wmma_f32_16x16x32_bf16(false, a10, false, bq, (short)0, acc[0], false, false);
      acc[1] = __builtin_amdgcn_wmma_f32_16x16x32_bf16(false, a11, false, bq, (short)0, acc[1], false, false);
      acc[2] = __builtin_amdgcn_wmma_f32_16x16x32_bf16(false, a20, false, bk, (short)0, acc[2], false, false);
      acc[3] = __builtin_amdgcn_wmma_f32_16x16x32_bf16(false, a21, false, bk, (short)0, acc[3], false, false);
      acc[4] = __builtin_amdgcn_wmma_f32_16x16x32_bf16(false, a20, false, bv, (short)0, acc[4], false, false);
      acc[5] = __builtin_amdgcn_wmma_f32_16x16x32_bf16(false, a21, false, bv, (short)0, acc[5], false, false);
    }
    __syncthreads();
  }
  // spill accumulators to sQKV as bf16 (C/D layout: M = rr + 8*hi, N = lo)
  #pragma unroll
  for (int j = 0; j < 6; ++j) {
    const int mat = j >> 1;
    const int tm  = j & 1;
    #pragma unroll
    for (int rr = 0; rr < 8; ++rr) {
      int mrow = tm * 16 + rr + hi * 8;
      sQKV[(mat * 32 + mrow) * 128 + w16 + lo] = (__bf16)acc[j][rr];
    }
  }
  __syncthreads();

  // ======== Phase B: per-row 4-head softmax attention (tiny) ========
  if (tid < 128) {
    int r = tid >> 2, h = tid & 3;
    const __bf16* qrow = sQKV + (0 * 32 + r) * 128 + h * 32;
    float lg[4];
    #pragma unroll
    for (int g = 0; g < 4; ++g) {
      const __bf16* krow = sQKV + (1 * 32 + r) * 128 + g * 32;
      float s = 0.f;
      #pragma unroll
      for (int jj = 0; jj < 32; ++jj) s += (float)qrow[jj] * (float)krow[jj];
      lg[g] = s * 0.17677669529663687f;      // 1/sqrt(32)
    }
    float mx = fmaxf(fmaxf(lg[0], lg[1]), fmaxf(lg[2], lg[3]));
    float e[4], ssum = 0.f;
    #pragma unroll
    for (int g = 0; g < 4; ++g) { e[g] = __expf(lg[g] - mx); ssum += e[g]; }
    float inv = 1.f / ssum;
    float o[32];
    #pragma unroll
    for (int jj = 0; jj < 32; ++jj) o[jj] = 0.f;
    #pragma unroll
    for (int g = 0; g < 4; ++g) {
      float a = e[g] * inv;
      const __bf16* vrow = sQKV + (2 * 32 + r) * 128 + g * 32;
      #pragma unroll
      for (int jj = 0; jj < 32; ++jj) o[jj] += a * (float)vrow[jj];
    }
    #pragma unroll
    for (int jj = 0; jj < 32; ++jj)
      sO[r * 128 + h * 32 + jj] = (__bf16)o[jj];
  }
  __syncthreads();

  // ======== Phase C: out = O @ Wo + bo  ([32,128]x[128,1024]) ========
  const int tm  = w >> 2;
  const int tnl = w & 3;
  for (int nc = 0; nc < IN_DIM / NC; ++nc) {
    if (useWs) {
      const __bf16* src = (const __bf16*)wsWo_ + (size_t)nc * (NC * 128);
      uint32_t ldsBase = (uint32_t)(uintptr_t)sWo;
      #pragma unroll
      for (int i = 0; i < 4; ++i) {
        uint32_t boff = (uint32_t)(tid + i * 256) * 16u;   // 1024 x b128
        async_b128(ldsBase + boff, boff, src);
      }
      wait_async0();
    } else {
      #pragma unroll
      for (int i = 0; i < 8; ++i) {
        int idx = tid + i * 256;               // 2048 float4
        int kk = idx >> 4;
        int n4 = idx & 15;
        float4 wv = *(const float4*)(Wo + (size_t)kk * IN_DIM + nc * NC + n4 * 4);
        sWo[(n4 * 4 + 0) * 128 + kk] = (__bf16)wv.x;
        sWo[(n4 * 4 + 1) * 128 + kk] = (__bf16)wv.y;
        sWo[(n4 * 4 + 2) * 128 + kk] = (__bf16)wv.z;
        sWo[(n4 * 4 + 3) * 128 + kk] = (__bf16)wv.w;
      }
    }
    __syncthreads();
    v8f c = {};
    #pragma unroll
    for (int ks = 0; ks < 4; ++ks) {
      v16bf a = load_a_frag(sO, 128, tm * 16, ks * 32, lane);
      v16bf b = load_b_frag(sWo, 128, tnl * 16, ks * 32, lane);
      c = __builtin_amdgcn_wmma_f32_16x16x32_bf16(
          false, a, false, b, (short)0, c, false, false);
    }
    int colg = nc * NC + tnl * 16 + lo;
    float bias = bo[colg];
    #pragma unroll
    for (int rr = 0; rr < 8; ++rr) {
      int grow = row0 + tm * 16 + rr + hi * 8;
      out[(size_t)grow * IN_DIM + colg] = c[rr] + bias;
    }
    __syncthreads();
  }
}

extern "C" void kernel_launch(void* const* d_in, const int* in_sizes, int n_in,
                              void* d_out, int out_size, void* d_ws, size_t ws_size,
                              hipStream_t stream) {
  (void)in_sizes; (void)n_in; (void)out_size;
  const float* x1 = (const float*)d_in[0];
  const float* x2 = (const float*)d_in[1];
  const float* Wq = (const float*)d_in[2];
  const float* Wk = (const float*)d_in[3];
  const float* Wv = (const float*)d_in[4];
  const float* Wo = (const float*)d_in[5];
  const float* bo = (const float*)d_in[6];
  float* out = (float*)d_out;

  unsigned short* wsW  = (unsigned short*)d_ws;
  unsigned short* wsWo = wsW + WSW_ELEMS;
  const int useWs = (d_ws != nullptr && ws_size >= WS_BYTES) ? 1 : 0;

  if (useWs) {
    prep_weights<<<dim3((WSW_ELEMS + WSWO_ELEMS) / 256), dim3(256), 0, stream>>>(
        Wq, Wk, Wv, Wo, wsW, wsWo);
  }
  const int blocks = 32768 / M_TILE;   // 1024 blocks x 256 threads (8 waves)
  xattn_fused<<<dim3(blocks), dim3(256), 0, stream>>>(
      x1, x2, Wq, Wk, Wv, Wo, bo, out, wsW, wsWo, useWs);
}
